// Attention_15599321219445
// MI455X (gfx1250) — matrix operand order repaired and checked by reading, not compiled, not run on hardware
//
#include <hip/hip_runtime.h>
#include <hip/hip_bf16.h>
#include <math.h>

// ---------------------------------------------------------------------------
// Bahdanau attention scores for MI455X (gfx1250, wave32, WMMA).
//   scores[s,b] = sum_h v[h] * tanh( enc[s,b,:]·W2[h,:] + hidden[b,:]·W1[h,:] + b[h] )
//   out[b,1,s]  = softmax_s(scores[s,b])
// Big GEMM (137 GFLOP) runs on v_wmma_f32_16x16x32_bf16 with f32 accumulate.
// ---------------------------------------------------------------------------

typedef __attribute__((ext_vector_type(16))) __bf16 v16bf;
typedef __attribute__((ext_vector_type(8)))  __bf16 v8bf;
typedef __attribute__((ext_vector_type(8)))  float  v8f;

#define S_LEN 2048
#define B_SZ  32
#define H_SZ  1024
#define M_TOT (S_LEN * B_SZ)   // 65536 rows of the fused GEMM

// Hardware TANH trans op on gfx1250 (keeps EXEC uniform; no libm branch).
__device__ __forceinline__ float fast_tanh(float x) {
#if __has_builtin(__builtin_amdgcn_tanhf)
  return __builtin_amdgcn_tanhf(x);
#else
  return tanhf(x);
#endif
}

// ---------------------------------------------------------------------------
// Kernel 1: hidT[n][b] = b_attn[n] + sum_k hidden[b,k] * W_attn[n,k]   (k < H)
// grid 1024, block 128. Trivial cost (67 MFLOP).
// ---------------------------------------------------------------------------
__global__ __launch_bounds__(128) void hid_prep_kernel(
    const float* __restrict__ hidden,   // [32,1024]
    const float* __restrict__ Wattn,    // [1024,2048]
    const float* __restrict__ battn,    // [1024]
    float* __restrict__ hidT)           // [1024,32]
{
  __shared__ float wrow[H_SZ];
  const int n = blockIdx.x;
  const float* wr = Wattn + (size_t)n * (2 * H_SZ);
  for (int k = threadIdx.x; k < H_SZ; k += blockDim.x) wrow[k] = wr[k];
  __syncthreads();
  if (threadIdx.x < B_SZ) {
    const int b = threadIdx.x;
    const float* hrow = hidden + (size_t)b * H_SZ;
    float s = 0.0f;
    for (int k = 0; k < H_SZ; ++k) s = fmaf(hrow[k], wrow[k], s);
    hidT[n * B_SZ + b] = s + battn[n];
  }
}

// ---------------------------------------------------------------------------
// Kernel 2: pack W2 = W_attn[:, H:2H] into bf16 WMMA B-operand layout.
// Flat pack index = ((t*32 + kb)*32 + lane)*16 + slot, where within one
// 32-K block: lane = 16*(k16) + n_local, slot = k%16   (k16 = (k%32)/16).
// ---------------------------------------------------------------------------
__global__ __launch_bounds__(256) void pack_w2_kernel(
    const float* __restrict__ Wattn,    // [1024,2048]
    __bf16* __restrict__ w2pack)        // [1024*1024] bf16
{
  const int idx  = blockIdx.x * 256 + threadIdx.x;   // 0 .. 1M-1
  const int slot = idx & 15;
  const int ln   = (idx >> 4) & 31;
  const int kb   = (idx >> 9) & 31;
  const int t    = idx >> 14;                        // N tile 0..63
  const int n    = (t << 4) + (ln & 15);
  const int k    = (kb << 5) + ((ln >> 4) << 4) + slot;
  const float w  = Wattn[(size_t)n * (2 * H_SZ) + H_SZ + k];
  w2pack[idx] = (__bf16)w;                           // native cvt, RNE
}

// ---------------------------------------------------------------------------
// Kernel 3: fused GEMM + tanh + v-dot.  grid = M_TOT/32 WGs of 256 threads
// (8 waves). M-tile = 32 rows = one s, all 32 b. A staged in LDS as bf16 in
// A-fragments, k = 16*(lane/16) + slot; B streamed from L2-resident w2pack.
// ---------------------------------------------------------------------------
__global__ __launch_bounds__(256) void attn_gemm_kernel(
    const float* __restrict__ enc,            // [65536,1024] f32
    const __bf16* __restrict__ w2pack,        // bf16 packed B
    const float* __restrict__ hidT,           // [1024,32]
    const float* __restrict__ vvec,           // [1024]
    float* __restrict__ scores)               // [65536] = [s][b]
{
  extern __shared__ __align__(16) char smem[];
  __bf16* Abuf = (__bf16*)smem;                   // 64KB: [sub2][kb32][lane32][slot16]
  float* s_scores = (float*)(smem + 65536);       // 32 floats

  const int tid  = threadIdx.x;
  const int lane = tid & 31;
  const int wave = tid >> 5;
  const int m0   = blockIdx.x * 32;

  if (tid < 32) s_scores[tid] = 0.0f;

  // ---- Stage A: 32 rows x 1024 K, f32 -> bf16 RNE, into WMMA-A layout ----
  // A-frag k order used here, same as w2pack (16x32): lane<16: slots0-15=K0-15;
  // lane>=16: K16-31.  8 consecutive K share one 16B LDS run.
  const float* encTile = enc + (size_t)m0 * H_SZ;
#pragma unroll
  for (int it = 0; it < 16; ++it) {
    const int o       = tid + (it << 8);   // octet id, 32 rows * 128 octets
    const int m_local = o >> 7;
    const int oct     = o & 127;
    const int kb      = oct >> 2;
    const int g       = oct & 3;           // K-octet group within 32-K block
    const int sub     = m_local >> 4;
    const int ln      = (m_local & 15) | ((g >> 1) << 4);
    const int slotb   = (g & 1) << 3;
    const float4* src = (const float4*)(encTile + m_local * H_SZ + (oct << 3));
    const float4 a = src[0];
    const float4 b = src[1];
    v8bf t;
    t[0] = (__bf16)a.x; t[1] = (__bf16)a.y; t[2] = (__bf16)a.z; t[3] = (__bf16)a.w;
    t[4] = (__bf16)b.x; t[5] = (__bf16)b.y; t[6] = (__bf16)b.z; t[7] = (__bf16)b.w;
    *(v8bf*)&Abuf[(((((sub << 5) + kb) << 5) + ln) << 4) + slotb] = t;
  }
  __syncthreads();

  float sacc0[8], sacc1[8];
#pragma unroll
  for (int r = 0; r < 8; ++r) { sacc0[r] = 0.0f; sacc1[r] = 0.0f; }

  // Each wave owns N-tiles t = wave*8 .. wave*8+7 (columns n = 16t..16t+15)
  for (int i = 0; i < 8; ++i) {
    const int t = (wave << 3) + i;
    v8f acc0 = {0.f, 0.f, 0.f, 0.f, 0.f, 0.f, 0.f, 0.f};
    v8f acc1 = acc0;
    const v16bf* Bp = (const v16bf*)(w2pack + (size_t)t * 32 * 32 * 16);
#pragma unroll 4
    for (int kb = 0; kb < 32; ++kb) {
      v16bf bfrag = Bp[(kb << 5) + lane];                                   // 32B/lane, coalesced
      v16bf a0 = *(const v16bf*)&Abuf[(((kb << 5) + lane) << 4)];           // sub 0
      v16bf a1 = *(const v16bf*)&Abuf[(((((32 + kb)) << 5) + lane) << 4)];  // sub 1
      acc0 = __builtin_amdgcn_wmma_f32_16x16x32_bf16(false, a0, false, bfrag,
                                                     (short)0, acc0, false, false);
      acc1 = __builtin_amdgcn_wmma_f32_16x16x32_bf16(false, a1, false, bfrag,
                                                     (short)0, acc1, false, false);
    }
    // Epilogue: D layout lane L, vgpr r -> M = r + 8*(L/16), N = L%16.
    const int nloc    = lane & 15;
    const int n       = (t << 4) + nloc;
    const float vn    = vvec[n];
    const int halfsel = (lane >> 4) << 3;   // 0 or 8
    {
      const float* hp = hidT + n * 32 + halfsel;         // rows b = halfsel+r (sub 0)
      const float4 h0 = *(const float4*)hp;
      const float4 h1 = *(const float4*)(hp + 4);
      const float hv[8] = {h0.x, h0.y, h0.z, h0.w, h1.x, h1.y, h1.z, h1.w};
#pragma unroll
      for (int r = 0; r < 8; ++r) sacc0[r] += vn * fast_tanh(acc0[r] + hv[r]);
    }
    {
      const float* hp = hidT + n * 32 + 16 + halfsel;    // rows b = 16+halfsel+r (sub 1)
      const float4 h0 = *(const float4*)hp;
      const float4 h1 = *(const float4*)(hp + 4);
      const float hv[8] = {h0.x, h0.y, h0.z, h0.w, h1.x, h1.y, h1.z, h1.w};
#pragma unroll
      for (int r = 0; r < 8; ++r) sacc1[r] += vn * fast_tanh(acc1[r] + hv[r]);
    }
  }

  // Reduce over n: butterfly within each 16-lane half, then LDS atomic add.
#pragma unroll
  for (int r = 0; r < 8; ++r) {
    float x0 = sacc0[r];
    float x1 = sacc1[r];
#pragma unroll
    for (int off = 1; off < 16; off <<= 1) {
      x0 += __shfl_xor(x0, off, 32);
      x1 += __shfl_xor(x1, off, 32);
    }
    if ((lane & 15) == 0) {
      const int row = ((lane >> 4) << 3) + r;
      atomicAdd(&s_scores[row], x0);        // ds_add_f32
      atomicAdd(&s_scores[16 + row], x1);
    }
  }
  __syncthreads();
  if (tid < 32) scores[m0 + tid] = s_scores[tid];
}

// ---------------------------------------------------------------------------
// Kernel 4: softmax over S per batch row. grid 32, block 256.
// out[b,0,s] = softmax_s(scores[s,b])
// ---------------------------------------------------------------------------
__global__ __launch_bounds__(256) void softmax_kernel(
    const float* __restrict__ scores, float* __restrict__ out)
{
  __shared__ float red[8];
  __shared__ float bc;
  const int b = blockIdx.x, tid = threadIdx.x, lane = tid & 31, wv = tid >> 5;
  float x[8];
#pragma unroll
  for (int i = 0; i < 8; ++i) x[i] = scores[(tid + i * 256) * B_SZ + b];
  float mx = x[0];
#pragma unroll
  for (int i = 1; i < 8; ++i) mx = fmaxf(mx, x[i]);
  for (int off = 16; off > 0; off >>= 1) mx = fmaxf(mx, __shfl_xor(mx, off, 32));
  if (lane == 0) red[wv] = mx;
  __syncthreads();
  if (tid == 0) {
    float m = red[0];
    for (int i = 1; i < 8; ++i) m = fmaxf(m, red[i]);
    bc = m;
  }
  __syncthreads();
  mx = bc;
  float sum = 0.0f;
#pragma unroll
  for (int i = 0; i < 8; ++i) { x[i] = expf(x[i] - mx); sum += x[i]; }
  for (int off = 16; off > 0; off >>= 1) sum += __shfl_xor(sum, off, 32);
  __syncthreads();
  if (lane == 0) red[wv] = sum;
  __syncthreads();
  if (tid == 0) {
    float s = 0.0f;
    for (int i = 0; i < 8; ++i) s += red[i];
    bc = s;
  }
  __syncthreads();
  const float inv = 1.0f / bc;
#pragma unroll
  for (int i = 0; i < 8; ++i) out[(size_t)b * S_LEN + tid + i * 256] = x[i] * inv;
}

// ---------------------------------------------------------------------------
extern "C" void kernel_launch(void* const* d_in, const int* in_sizes, int n_in,
                              void* d_out, int out_size, void* d_ws, size_t ws_size,
                              hipStream_t stream) {
  const float* hidden = (const float*)d_in[0];   // [1,32,1024]
  const float* enc    = (const float*)d_in[1];   // [2048,32,1024]
  const float* Wattn  = (const float*)d_in[2];   // [1024,2048]
  const float* battn  = (const float*)d_in[3];   // [1024]
  const float* vvec   = (const float*)d_in[4];   // [1024]
  float* out = (float*)d_out;                    // [32,1,2048]

  char* ws = (char*)d_ws;
  float*   hidT   = (float*)(ws);                           // 128 KB
  __bf16*  w2pack = (__bf16*)(ws + (128 << 10));            // 2 MB
  float*   scores = (float*)(ws + (128 << 10) + (2 << 20)); // 256 KB
  // total ws usage: ~2.4 MB

  hid_prep_kernel<<<H_SZ, 128, 0, stream>>>(hidden, Wattn, battn, hidT);
  pack_w2_kernel<<<(H_SZ * H_SZ) / 256, 256, 0, stream>>>(Wattn, w2pack);
  attn_gemm_kernel<<<M_TOT / 32, 256, 65536 + 128, stream>>>(enc, w2pack, hidT,
                                                             vvec, scores);
  softmax_kernel<<<B_SZ, 256, 0, stream>>>(scores, out);
}
